// MCMambaBlock_31215822307446
// MI455X (gfx1250) — matrix operand order repaired and checked by reading, not compiled, not run on hardware
//
#include <hip/hip_runtime.h>
#include <hip/hip_bf16.h>

// MCMamba block for MI455X (gfx1250): TDM-staged WMMA f16 GEMMs + lane-split scan.
#define B_      2
#define T_      1024
#define DM_     768
#define DI_     1536
#define DS_     64
#define NP_     256     // x_proj output padded 160 -> 256 (multiple of 128 tile)
#define SEG_    256
#define NSEG_   4
#define DTRANK_ 32

typedef __attribute__((ext_vector_type(16))) _Float16 v16h;
typedef __attribute__((ext_vector_type(8)))  float    v8f;
typedef unsigned int u32x4 __attribute__((ext_vector_type(4)));
typedef int          i32x8 __attribute__((ext_vector_type(8)));
typedef int          i32x4 __attribute__((ext_vector_type(4)));

// ---------------- conversions ----------------
__global__ void k_f32_to_f16(const float* __restrict__ s, _Float16* __restrict__ d, int n) {
    int i = blockIdx.x * 256 + threadIdx.x;
    if (i < n) d[i] = (_Float16)s[i];
}

__global__ void k_f32_to_f16_pad(const float* __restrict__ s, _Float16* __restrict__ d,
                                 int rows, int cols, int ldd) {
    int i = blockIdx.x * 256 + threadIdx.x;
    if (i >= rows * ldd) return;
    int r = i / ldd, c = i - r * ldd;
    d[i] = (c < cols) ? (_Float16)s[r * cols + c] : (_Float16)0.0f;
}

// ---------------- TDM: DMA one 2D f16 tile (width x height) into LDS ----------------
// D# per CDNA5 ISA. data_size=1 (2 bytes), no padding,
// no gather, no multicast (workgroup_mask=0), 2D tile (tile_dim2=0), groups 2/3 zero.
__device__ __forceinline__ void tdm_load_2d(const _Float16* gptr, unsigned lds_off,
                                            unsigned width, unsigned height,
                                            unsigned stride_elems) {
    unsigned long long ga = (unsigned long long)(uintptr_t)gptr;
    unsigned td0 = stride_elems;       // generous tensor_dim0 (>= tile width, no OOB hit)
    unsigned td1 = 1u << 20;           // generous tensor_dim1
    u32x4 g0;
    g0[0] = 1u;                                            // count=1, user mode
    g0[1] = lds_off;                                       // lds_addr (bytes)
    g0[2] = (unsigned)ga;                                  // global_addr[31:0]
    g0[3] = (unsigned)((ga >> 32) & 0x1FFFFFFu) | (2u << 30); // addr[56:32] | type=2
    i32x8 g1;
    g1[0] = (int)(1u << 16);                               // data_size=2B, mask=0, flags=0
    g1[1] = (int)((td0 & 0xFFFFu) << 16);                  // tensor_dim0[15:0] @ bits63:48
    g1[2] = (int)((td0 >> 16) | ((td1 & 0xFFFFu) << 16));  // dim0[31:16] | dim1[15:0]
    g1[3] = (int)((td1 >> 16) | (width << 16));            // dim1[31:16] | tile_dim0
    g1[4] = (int)height;                                   // tile_dim1 | tile_dim2=0
    g1[5] = (int)stride_elems;                             // tensor_dim0_stride[31:0]
    g1[6] = 0;                                             // stride0[47:32] | stride1[15:0]
    g1[7] = 0;                                             // stride1[47:16]
    i32x4 g2 = {0, 0, 0, 0};
    i32x4 g3 = {0, 0, 0, 0};
#if defined(__clang_major__) && __clang_major__ >= 23
    i32x8 g4 = {0, 0, 0, 0, 0, 0, 0, 0};
    __builtin_amdgcn_tensor_load_to_lds(g0, g1, g2, g3, g4, 0);
#else
    __builtin_amdgcn_tensor_load_to_lds(g0, g1, g2, g3, 0);
#endif
}

// ---------------- WMMA GEMM with TDM double-buffered LDS staging ----------------
// C[f32, MxN] = A[f16, MxK] @ W[f16, KxN]. grid=(N/128, M/64), block=256 (8 waves).
// Per k-tile (K step 32): A-tile 64x32 (wave0 DMA), B-tile 32x128 (wave1 DMA).
// Wave w: rows (w>>1)*16, cols (w&1)*64 -> 4x v_wmma_f32_16x16x32_f16 per tile.
#define TM_    64
#define TN_    128
#define ATILE_ (TM_ * 32)          // 2048 f16 = 4KB
#define BTILE_ (32 * TN_)          // 4096 f16 = 8KB
#define TBUF_  (ATILE_ + BTILE_)   // 12KB per buffer, x2 = 24KB LDS

__global__ void k_wmma_gemm(const _Float16* __restrict__ A, const _Float16* __restrict__ W,
                            float* __restrict__ C, int K, int lda, int ldw, int ldc) {
    __shared__ __align__(16) _Float16 ldsbuf[2][TBUF_];

    int lane = threadIdx.x & 31;
    int wave = threadIdx.x >> 5;
    int rw = wave >> 1, cw = wave & 1;
    int m0 = blockIdx.y * TM_;
    int n0 = blockIdx.x * TN_;
    int KT = K >> 5;

    // Uniform LDS byte base for the TDM descriptors.
    unsigned lds_base = (unsigned)(unsigned long long)(uintptr_t)&ldsbuf[0][0];

    v8f acc[4] = {{}, {}, {}, {}};

    if (wave == 0)
        tdm_load_2d(A + (size_t)m0 * lda, lds_base,              32, TM_, (unsigned)lda);
    else if (wave == 1)
        tdm_load_2d(W + n0,              lds_base + ATILE_ * 2, TN_, 32,  (unsigned)ldw);

    // Fragment offsets (ISA 7.12.2):
    //  A 16x32: lane<16 row=lane K{0..7,16..23}; lane>=16 row=lane-16 K{8..15,24..31}
    //  B 32x16: lane<16 col=lane K{0..7,16..23}; lane>=16 col=lane-16 K{8..15,24..31}
    int a_off = (rw * 16 + (lane & 15)) * 32 + ((lane >> 4) << 3);
    int b_off = ATILE_ + ((lane >> 4) << 3) * TN_ + (lane & 15) + cw * 64;

    // Opaque zero (SGPR): keeps AS3 intact but stops undef-folding of LDS reads.
    unsigned zoff = 0;
    asm volatile("" : "+s"(zoff));

    for (int kt = 0; kt < KT; ++kt) {
        // Per-wave TENSORcnt: no-op for non-issuing waves; issuing waves drain their
        // DMA *before* signaling, so the barrier publishes the tile to everyone.
        __builtin_amdgcn_s_wait_tensorcnt(0);
        __syncthreads();
        asm volatile("" ::: "memory");      // TDM wrote LDS behind the compiler's back
        int buf = kt & 1;
        if (kt + 1 < KT) {                   // overlap DMA(kt+1) with compute(kt)
            int k1 = (kt + 1) << 5;
            unsigned nb = lds_base + (unsigned)(((kt + 1) & 1) * TBUF_ * 2);
            if (wave == 0)
                tdm_load_2d(A + (size_t)m0 * lda + k1, nb,              32, TM_, (unsigned)lda);
            else if (wave == 1)
                tdm_load_2d(W + (size_t)k1 * ldw + n0, nb + ATILE_ * 2, TN_, 32, (unsigned)ldw);
        }
        const _Float16* ab = &ldsbuf[buf][a_off + zoff];
        v16h a;
        ((float4*)&a)[0] = *(const float4*)(ab);        // ds_load_b128
        ((float4*)&a)[1] = *(const float4*)(ab + 16);
#pragma unroll
        for (int j = 0; j < 4; ++j) {
            const _Float16* bb = &ldsbuf[buf][b_off + j * 16 + zoff];
            v16h bf;
#pragma unroll
            for (int s = 0; s < 8; ++s) {
                bf[s] = bb[s * TN_];
                bf[s + 8] = bb[(s + 16) * TN_];
            }
            acc[j] = __builtin_amdgcn_wmma_f32_16x16x32_f16(false, a, false, bf,
                                                            (short)0, acc[j], false, false);
        }
        __syncthreads();                     // reads done before this buffer is re-DMAed
    }

    // C/D layout: VGPR r -> row +8*(lane>=16)+r, col = lane%16.
    int crow = m0 + rw * 16 + ((lane >> 4) << 3);
#pragma unroll
    for (int j = 0; j < 4; ++j) {
        int ccol = n0 + cw * 64 + j * 16 + (lane & 15);
#pragma unroll
        for (int rr = 0; rr < 8; ++rr)
            C[(size_t)(crow + rr) * ldc + ccol] = acc[j][rr];
    }
}

// ---------------- depthwise causal conv (k=4) + bias + silu ----------------
__global__ void k_conv_silu(const float* __restrict__ xz, const float* __restrict__ cw,
                            const float* __restrict__ cb, float* __restrict__ xi,
                            _Float16* __restrict__ xih) {
    int i = blockIdx.x * 256 + threadIdx.x;
    if (i >= B_ * T_ * DI_) return;
    int d = i % DI_;
    int bt = i / DI_;
    int t = bt % T_;
    float acc = cb[d];
#pragma unroll
    for (int k = 0; k < 4; ++k) {
        int tt = t - 3 + k;
        if (tt >= 0) acc += cw[d * 4 + k] * xz[(size_t)(bt - t + tt) * (2 * DI_) + d];
    }
    float sv = acc / (1.0f + __expf(-acc));
    xi[i] = sv;
    xih[i] = (_Float16)sv;
}

// ---------------- dt extraction / softplus epilogue ----------------
__global__ void k_extract_dt(const float* __restrict__ xdbl, _Float16* __restrict__ dth) {
    int i = blockIdx.x * 256 + threadIdx.x;
    if (i >= B_ * T_ * DTRANK_) return;
    int r = i >> 5, c = i & 31;
    dth[i] = (_Float16)xdbl[(size_t)r * NP_ + c];
}

__global__ void k_softplus_bias(float* __restrict__ dt, const float* __restrict__ bias) {
    int i = blockIdx.x * 256 + threadIdx.x;
    if (i >= B_ * T_ * DI_) return;
    float v = dt[i] + bias[i % DI_];
    dt[i] = (v > 20.0f) ? v : logf(1.0f + __expf(v));
}

// ---------------- selective scan ----------------
// 16 lanes per channel d, 4 states per lane (wave32: two channel-groups per wave).
__global__ void k_scan(const float* __restrict__ dtf, const float* __restrict__ xdbl,
                       const float* __restrict__ xi, const float* __restrict__ xz,
                       const float* __restrict__ A_log, const float* __restrict__ Dp,
                       _Float16* __restrict__ ygh) {
    int lane16 = threadIdx.x & 15;
    int grp    = threadIdx.x >> 4;
    int d = blockIdx.x * 16 + grp;
    int b = blockIdx.y;
    int s0 = lane16 * 4;

    float A0 = -__expf(A_log[(size_t)d * DS_ + s0 + 0]);
    float A1 = -__expf(A_log[(size_t)d * DS_ + s0 + 1]);
    float A2 = -__expf(A_log[(size_t)d * DS_ + s0 + 2]);
    float A3 = -__expf(A_log[(size_t)d * DS_ + s0 + 3]);
    float Dd = Dp[d];
    float h0 = 0.f, h1 = 0.f, h2 = 0.f, h3 = 0.f;

    for (int t = 0; t < T_; ++t) {
        size_t r = (size_t)(b * T_ + t);
        float dtv = dtf[r * DI_ + d];
        float uv  = xi[r * DI_ + d];
        float du  = dtv * uv;
        const float* xr = xdbl + r * NP_;
        float Bv0 = xr[32 + s0], Bv1 = xr[33 + s0], Bv2 = xr[34 + s0], Bv3 = xr[35 + s0];
        float Cv0 = xr[96 + s0], Cv1 = xr[97 + s0], Cv2 = xr[98 + s0], Cv3 = xr[99 + s0];

        h0 = __expf(dtv * A0) * h0 + du * Bv0;
        h1 = __expf(dtv * A1) * h1 + du * Bv1;
        h2 = __expf(dtv * A2) * h2 + du * Bv2;
        h3 = __expf(dtv * A3) * h3 + du * Bv3;

        float y = h0 * Cv0 + h1 * Cv1 + h2 * Cv2 + h3 * Cv3;
        y += __shfl_xor(y, 1, 32);
        y += __shfl_xor(y, 2, 32);
        y += __shfl_xor(y, 4, 32);
        y += __shfl_xor(y, 8, 32);

        if (lane16 == 0) {
            float yt = y + uv * Dd;
            float zv = xz[r * (2 * DI_) + DI_ + d];
            float g  = zv / (1.0f + __expf(-zv));
            ygh[r * DI_ + d] = (_Float16)(yt * g);
        }
    }
}

// ---------------- cumsum over T (per b,d) + segment means ----------------
__global__ void k_cumsum(const float* __restrict__ x, float* __restrict__ cs,
                         float* __restrict__ mst) {
    int i = blockIdx.x * 256 + threadIdx.x;
    if (i >= B_ * DM_) return;
    int b = i / DM_, d = i - b * DM_;
    float run = 0.0f, prev = 0.0f;
    for (int t = 0; t < T_; ++t) {
        run += x[(size_t)(b * T_ + t) * DM_ + d];
        cs[(size_t)(b * T_ + t) * DM_ + d] = run;
        if ((t & (SEG_ - 1)) == (SEG_ - 1)) {
            int j = t / SEG_;
            mst[((size_t)b * NSEG_ + j) * DM_ + d] = (run - prev) * (1.0f / SEG_);
            prev = run;
        }
    }
}

// ---------------- retrieval: 5-way softmax gate + combine ----------------
__global__ void k_retrieve(const float* __restrict__ u, const float* __restrict__ cs,
                           const float* __restrict__ mst, const float* __restrict__ xout,
                           float* __restrict__ out) {
    int t = blockIdx.x, b = blockIdx.y;
    int seg = t / SEG_;
    size_t r = (size_t)(b * T_ + t);
    float p[5] = {0.f, 0.f, 0.f, 0.f, 0.f};
    float inv_len = 1.0f / (float)(t - seg * SEG_ + 1);
    size_t baserow = (seg > 0) ? (size_t)(b * T_ + seg * SEG_ - 1) : 0;

    for (int d = threadIdx.x; d < DM_; d += 256) {
        float uv = u[r * DM_ + d];
#pragma unroll
        for (int j = 0; j < NSEG_; ++j)
            p[j] += uv * mst[((size_t)b * NSEG_ + j) * DM_ + d];
        float base = (seg > 0) ? cs[baserow * DM_ + d] : 0.0f;
        p[4] += uv * (cs[r * DM_ + d] - base) * inv_len;
    }
#pragma unroll
    for (int j = 0; j < 5; ++j)
        for (int m = 1; m < 32; m <<= 1)
            p[j] += __shfl_xor(p[j], m, 32);

    __shared__ float sred[8][5];
    __shared__ float gates[5];
    int wave = threadIdx.x >> 5;
    if ((threadIdx.x & 31) == 0)
        for (int j = 0; j < 5; ++j) sred[wave][j] = p[j];
    __syncthreads();

    if (threadIdx.x == 0) {
        const float scale = 0.03608439182435161f; // 1/sqrt(768)
        float sc[5];
        for (int j = 0; j < 5; ++j) {
            float v = 0.f;
            for (int w2 = 0; w2 < 8; ++w2) v += sred[w2][j];
            sc[j] = v * scale;
        }
        float mx = sc[4];
        for (int j = 0; j < NSEG_; ++j)
            if (j < seg && sc[j] > mx) mx = sc[j];
        float e[5], se = 0.f;
        for (int j = 0; j < NSEG_; ++j) {
            e[j] = (j < seg) ? __expf(sc[j] - mx) : 0.0f;   // causal mask
            se += e[j];
        }
        e[4] = __expf(sc[4] - mx);
        se += e[4];
        for (int j = 0; j < 5; ++j) gates[j] = e[j] / se;
    }
    __syncthreads();

    for (int d = threadIdx.x; d < DM_; d += 256) {
        float acc = gates[4] * xout[r * DM_ + d];
        for (int j = 0; j < NSEG_; ++j)
            if (j < seg)
                acc += gates[j] * xout[(size_t)(b * T_ + (j + 1) * SEG_ - 1) * DM_ + d];
        out[r * DM_ + d] = acc;
    }
}

// ---------------- host launcher ----------------
extern "C" void kernel_launch(void* const* d_in, const int* in_sizes, int n_in,
                              void* d_out, int out_size, void* d_ws, size_t ws_size,
                              hipStream_t stream) {
    const float* x    = (const float*)d_in[0];
    const float* wi   = (const float*)d_in[1];
    const float* cw   = (const float*)d_in[2];
    const float* cb   = (const float*)d_in[3];
    const float* wxp  = (const float*)d_in[4];
    const float* wdt  = (const float*)d_in[5];
    const float* bdt  = (const float*)d_in[6];
    const float* alog = (const float*)d_in[7];
    const float* dpar = (const float*)d_in[8];
    const float* wout = (const float*)d_in[9];
    const float* wu   = (const float*)d_in[10];
    float* out = (float*)d_out;

    char* base = (char*)d_ws;
    size_t off = 0;
    auto alloc = [&](size_t bytes) -> void* {
        void* p = base + off;
        off += (bytes + 255) & ~(size_t)255;
        return p;
    };
    const size_t BT = (size_t)B_ * T_;
    _Float16* Xh    = (_Float16*)alloc(BT * DM_ * 2);
    _Float16* WinH  = (_Float16*)alloc((size_t)DM_ * 2 * DI_ * 2);
    float*    XZ    = (float*)   alloc(BT * 2 * DI_ * 4);
    float*    XI    = (float*)   alloc(BT * DI_ * 4);
    _Float16* XIh   = (_Float16*)alloc(BT * DI_ * 2);
    _Float16* WxpH  = (_Float16*)alloc((size_t)DI_ * NP_ * 2);
    float*    XDBL  = (float*)   alloc(BT * NP_ * 4);
    _Float16* DTh   = (_Float16*)alloc(BT * DTRANK_ * 2);
    _Float16* WdtH  = (_Float16*)alloc((size_t)DTRANK_ * DI_ * 2);
    float*    DTF   = (float*)   alloc(BT * DI_ * 4);
    _Float16* WoutH = (_Float16*)alloc((size_t)DI_ * DM_ * 2);
    _Float16* YGh   = (_Float16*)alloc(BT * DI_ * 2);
    float*    XOUT  = (float*)   alloc(BT * DM_ * 4);
    _Float16* WuH   = (_Float16*)alloc((size_t)DM_ * DM_ * 2);
    float*    Uu    = (float*)   alloc(BT * DM_ * 4);
    float*    CS    = (float*)   alloc(BT * DM_ * 4);
    float*    MST   = (float*)   alloc((size_t)B_ * NSEG_ * DM_ * 4);
    (void)ws_size; (void)n_in; (void)in_sizes; (void)out_size;

    auto cdiv = [](size_t a, size_t b) { return (int)((a + b - 1) / b); };

    // f32 -> f16 staging
    k_f32_to_f16<<<cdiv(BT * DM_, 256), 256, 0, stream>>>(x, Xh, (int)(BT * DM_));
    k_f32_to_f16<<<cdiv((size_t)DM_ * 2 * DI_, 256), 256, 0, stream>>>(wi, WinH, DM_ * 2 * DI_);
    k_f32_to_f16<<<cdiv((size_t)DTRANK_ * DI_, 256), 256, 0, stream>>>(wdt, WdtH, DTRANK_ * DI_);
    k_f32_to_f16<<<cdiv((size_t)DI_ * DM_, 256), 256, 0, stream>>>(wout, WoutH, DI_ * DM_);
    k_f32_to_f16<<<cdiv((size_t)DM_ * DM_, 256), 256, 0, stream>>>(wu, WuH, DM_ * DM_);
    k_f32_to_f16_pad<<<cdiv((size_t)DI_ * NP_, 256), 256, 0, stream>>>(
        wxp, WxpH, DI_, DTRANK_ + 2 * DS_, NP_);

    // xz = x @ in_proj_w               (2048 x 3072, K=768)
    k_wmma_gemm<<<dim3((2 * DI_) / TN_, (int)(BT / TM_)), 256, 0, stream>>>(
        Xh, WinH, XZ, DM_, DM_, 2 * DI_, 2 * DI_);
    // depthwise conv + silu
    k_conv_silu<<<cdiv(BT * DI_, 256), 256, 0, stream>>>(XZ, cw, cb, XI, XIh);
    // xdbl = xi @ x_proj_w             (2048 x 256pad, K=1536)
    k_wmma_gemm<<<dim3(NP_ / TN_, (int)(BT / TM_)), 256, 0, stream>>>(
        XIh, WxpH, XDBL, DI_, DI_, NP_, NP_);
    // dt_full = softplus(dt @ dt_proj_w + b)
    k_extract_dt<<<cdiv(BT * DTRANK_, 256), 256, 0, stream>>>(XDBL, DTh);
    k_wmma_gemm<<<dim3(DI_ / TN_, (int)(BT / TM_)), 256, 0, stream>>>(
        DTh, WdtH, DTF, DTRANK_, DTRANK_, DI_, DI_);
    k_softplus_bias<<<cdiv(BT * DI_, 256), 256, 0, stream>>>(DTF, bdt);
    // selective scan (fused D-skip, silu(z) gate, f16 output)
    k_scan<<<dim3(DI_ / 16, B_), 256, 0, stream>>>(DTF, XDBL, XI, XZ, alog, dpar, YGh);
    // x_out = y_gated @ out_proj_w     (2048 x 768, K=1536)
    k_wmma_gemm<<<dim3(DM_ / TN_, (int)(BT / TM_)), 256, 0, stream>>>(
        YGh, WoutH, XOUT, DI_, DI_, DM_, DM_);
    // u = x @ W_u                      (2048 x 768, K=768)
    k_wmma_gemm<<<dim3(DM_ / TN_, (int)(BT / TM_)), 256, 0, stream>>>(
        Xh, WuH, Uu, DM_, DM_, DM_, DM_);
    // retrieval path
    k_cumsum<<<cdiv((size_t)B_ * DM_, 256), 256, 0, stream>>>(x, CS, MST);
    k_retrieve<<<dim3(T_, B_), 256, 0, stream>>>(Uu, CS, MST, XOUT, out);
}
